// MyModel_87522843558860
// MI455X (gfx1250) — matrix operand order repaired and checked by reading, not compiled, not run on hardware
//
#include <hip/hip_runtime.h>
#include <hip/hip_bf16.h>

// MI455X / gfx1250 recurrent-BN LSTM.
// Phase 1: big bf16 WMMA GEMM  xg = x @ kernel  (+ fused BN rows 0-3 + bias).
// Phase 2: 256 sequential fused steps: zg = h @ recurrent_kernel (bf16 WMMA,
//          fp32 accum) + BN rows 4-7 + hard-sigmoid/tanh gate math + cell BN.
// A-tiles are staged with CDNA5 async global->LDS copies (ASYNCcnt path);
// B-tiles transpose in flight so they use the VGPR path.
// Workspace assumption: ws_size >= ~584 MB (gate pre-activations dominate).

typedef __attribute__((ext_vector_type(16))) __bf16 v16bf;
typedef __attribute__((ext_vector_type(8)))  float  v8f;

union FragAB { v16bf v; uint4 q[2]; };

constexpr int Bc = 128, Tc = 256, Dc = 512, Uc = 1024, G4 = 4096;
constexpr float BN_EPS = 1e-3f;

__device__ __forceinline__ float hsig(float x) {
    return fminf(fmaxf(0.2f * x + 0.5f, 0.0f), 1.0f);
}
__device__ __forceinline__ float fast_tanh(float x) {
#if __has_builtin(__builtin_amdgcn_tanhf)
    return __builtin_amdgcn_tanhf(x);   // v_tanh_f32 on gfx1250
#else
    return tanhf(x);
#endif
}

// Async global->LDS 16-byte copy (CDNA5 ASYNC path, no VGPR data round-trip).
__device__ __forceinline__ void async_copy_b128(const __bf16* gsrc, __bf16* lds_dst) {
    unsigned lds_off = (unsigned)(uintptr_t)lds_dst;   // addr[31:0] = LDS offset
    asm volatile("global_load_async_to_lds_b128 %0, %1, off"
                 :: "v"(lds_off), "v"(gsrc) : "memory");
}
__device__ __forceinline__ void wait_async_zero() {
    asm volatile("s_wait_asynccnt 0x0" ::: "memory");
}

// ---------------------------------------------------------------- convert
__global__ void f32_to_bf16_kernel(const float* __restrict__ s,
                                   __bf16* __restrict__ d, int n) {
    int i = blockIdx.x * blockDim.x + threadIdx.x;
    int stride = gridDim.x * blockDim.x;
    for (; i < n; i += stride) d[i] = (__bf16)s[i];
}

// ---------------------------------------------------------------- init h,c
__global__ void init_state_kernel(float* __restrict__ c,
                                  __bf16* __restrict__ h, int n) {
    int i = blockIdx.x * blockDim.x + threadIdx.x;
    int stride = gridDim.x * blockDim.x;
    for (; i < n; i += stride) { c[i] = 0.0f; h[i] = (__bf16)0.0f; }
}

// ---------------------------------------------------------------- input GEMM
// gates[t][b][g] = BN_g(x@W)[m][g] + bias[g],  m = b*T + t
// Block tile 128(M) x 128(N), K-step 32. 8 waves, each 64x32 (4x2 frags).
__global__ __launch_bounds__(256) void gemm_input_kernel(
    const __bf16* __restrict__ X, const __bf16* __restrict__ W,
    const float* __restrict__ bn_beta, const float* __restrict__ bn_gamma,
    const float* __restrict__ bn_mean, const float* __restrict__ bn_var,
    const float* __restrict__ bias, float* __restrict__ gates) {
    __shared__ __align__(16) __bf16 As[128 * 40];   // [m][k] pad 40
    __shared__ __align__(16) __bf16 Bs[128 * 40];   // [n][k] (transposed) pad 40

    const int tid  = threadIdx.x;
    const int lane = tid & 31, wid = tid >> 5;
    const int wm = wid & 1, wn = wid >> 1;          // wave grid 2 x 4
    const int mTile = blockIdx.x * 128;
    const int nTile = blockIdx.y * 128;

    v8f acc[4][2];
#pragma unroll
    for (int i = 0; i < 4; i++)
#pragma unroll
        for (int j = 0; j < 2; j++)
#pragma unroll
            for (int r = 0; r < 8; r++) acc[i][j][r] = 0.0f;

    for (int kt = 0; kt < Dc; kt += 32) {
        // stage A tile 128x32: async global->LDS b128 copies
#pragma unroll
        for (int p = 0; p < 2; p++) {
            int idx = tid + p * 256;                 // 0..511
            int row = idx >> 2, c4 = idx & 3;
            async_copy_b128(X + (size_t)(mTile + row) * Dc + kt + c4 * 8,
                            As + row * 40 + c4 * 8);
        }
        // stage B tile 32x128, transposed in flight -> Bs[n][k] (VGPR path)
        const unsigned* W32 = (const unsigned*)W;
#pragma unroll
        for (int p = 0; p < 8; p++) {
            int u = tid + p * 256;                   // 0..2047
            int k = u >> 6, npair = u & 63;
            unsigned v = W32[((size_t)(kt + k) * G4 + nTile) / 2 + npair];
            Bs[(npair * 2 + 0) * 40 + k] = ((const __bf16*)&v)[0];
            Bs[(npair * 2 + 1) * 40 + k] = ((const __bf16*)&v)[1];
        }
        wait_async_zero();
        __syncthreads();

        const int kHalf = lane >> 4, r16 = lane & 15;
        FragAB a[4], b[2];
#pragma unroll
        for (int i = 0; i < 4; i++) {                // A: M=r16, K 0-15 | 16-31
            const __bf16* base = As + (wm * 64 + i * 16 + r16) * 40 + kHalf * 16;
            a[i].q[0] = *(const uint4*)(base);
            a[i].q[1] = *(const uint4*)(base + 8);
        }
#pragma unroll
        for (int j = 0; j < 2; j++) {                // B: N=r16, K 0-15 | 16-31
            const __bf16* base = Bs + (wn * 32 + j * 16 + r16) * 40 + kHalf * 16;
            b[j].q[0] = *(const uint4*)(base);
            b[j].q[1] = *(const uint4*)(base + 8);
        }
#pragma unroll
        for (int i = 0; i < 4; i++)
#pragma unroll
            for (int j = 0; j < 2; j++)
                acc[i][j] = __builtin_amdgcn_wmma_f32_16x16x32_bf16(
                    false, a[i].v, false, b[j].v, (short)0, acc[i][j], false, false);
        __syncthreads();
    }

    // epilogue: fused BN (rows 0-3) + bias, scatter to time-major gates[t][b][g]
    const int lane15 = lane & 15, hi = lane >> 4;
#pragma unroll
    for (int j = 0; j < 2; j++) {
        int n = nTile + wn * 32 + j * 16 + lane15;
        int gi = n >> 10, u = n & (Uc - 1);
        float sc = bn_gamma[gi * Uc + u] * rsqrtf(bn_var[gi * Uc + u] + BN_EPS);
        float sh = bn_beta[gi * Uc + u] - bn_mean[gi * Uc + u] * sc + bias[n];
#pragma unroll
        for (int i = 0; i < 4; i++)
#pragma unroll
            for (int r = 0; r < 8; r++) {
                int m = mTile + wm * 64 + i * 16 + hi * 8 + r;
                int t = m & (Tc - 1), bb = m >> 8;   // m = b*T + t
                gates[((size_t)t * Bc + bb) * G4 + n] = acc[i][j][r] * sc + sh;
            }
    }
}

// ---------------------------------------------------------------- LSTM step
// One workgroup per 32-unit slice: computes z for ALL 4 gates of its units
// (columns u, u+U, u+2U, u+3U) so the cell update is workgroup-local.
__global__ __launch_bounds__(256) void lstm_step_kernel(
    int t, const __bf16* __restrict__ Hin, const __bf16* __restrict__ RK,
    const float* __restrict__ bn_beta, const float* __restrict__ bn_gamma,
    const float* __restrict__ bn_mean, const float* __restrict__ bn_var,
    const float* __restrict__ gates, float* __restrict__ cbuf,
    __bf16* __restrict__ Hout, float* __restrict__ out) {
    __shared__ __align__(16) __bf16 As[128 * 40];     // h tile [b][k]
    __shared__ __align__(16) __bf16 Bs[128 * 40];     // rk tile [n_local][k]
    __shared__ __bf16 Zs[128 * 136];                  // post-BN z [b][gate*32+ul]

    const int tid  = threadIdx.x;
    const int lane = tid & 31, wid = tid >> 5;
    const int wm = wid & 1, wg = wid >> 1;            // wg = gate 0..3
    const int u0 = blockIdx.x * 32;

    v8f acc[4][2];
#pragma unroll
    for (int i = 0; i < 4; i++)
#pragma unroll
        for (int j = 0; j < 2; j++)
#pragma unroll
            for (int r = 0; r < 8; r++) acc[i][j][r] = 0.0f;

    for (int kt = 0; kt < Uc; kt += 32) {
        // stage A: h tile [128][32], async global->LDS
#pragma unroll
        for (int p = 0; p < 2; p++) {
            int idx = tid + p * 256;
            int row = idx >> 2, c4 = idx & 3;
            async_copy_b128(Hin + (size_t)row * Uc + kt + c4 * 8,
                            As + row * 40 + c4 * 8);
        }
        const unsigned* R32 = (const unsigned*)RK;    // stage B: 4 gate slabs
#pragma unroll
        for (int p = 0; p < 8; p++) {
            int u = tid + p * 256;                    // 0..2047
            int k = u >> 6, rem = u & 63, g = rem >> 4, pr = rem & 15;
            unsigned v = R32[((size_t)(kt + k) * G4 + g * Uc + u0) / 2 + pr];
            Bs[(g * 32 + pr * 2 + 0) * 40 + k] = ((const __bf16*)&v)[0];
            Bs[(g * 32 + pr * 2 + 1) * 40 + k] = ((const __bf16*)&v)[1];
        }
        wait_async_zero();
        __syncthreads();

        const int kHalf = lane >> 4, r16 = lane & 15;
        FragAB a[4], b[2];
#pragma unroll
        for (int i = 0; i < 4; i++) {
            const __bf16* base = As + (wm * 64 + i * 16 + r16) * 40 + kHalf * 16;
            a[i].q[0] = *(const uint4*)(base);
            a[i].q[1] = *(const uint4*)(base + 8);
        }
#pragma unroll
        for (int j = 0; j < 2; j++) {
            const __bf16* base = Bs + (wg * 32 + j * 16 + r16) * 40 + kHalf * 16;
            b[j].q[0] = *(const uint4*)(base);
            b[j].q[1] = *(const uint4*)(base + 8);
        }
#pragma unroll
        for (int i = 0; i < 4; i++)
#pragma unroll
            for (int j = 0; j < 2; j++)
                acc[i][j] = __builtin_amdgcn_wmma_f32_16x16x32_bf16(
                    false, a[i].v, false, b[j].v, (short)0, acc[i][j], false, false);
        __syncthreads();
    }

    // fused BN rows 4..7, stash z in LDS
    const int lane15 = lane & 15, hi = lane >> 4;
#pragma unroll
    for (int j = 0; j < 2; j++) {
        int ul = j * 16 + lane15;
        int bi = (4 + wg) * Uc + u0 + ul;
        float sc = bn_gamma[bi] * rsqrtf(bn_var[bi] + BN_EPS);
        float sh = bn_beta[bi] - bn_mean[bi] * sc;
#pragma unroll
        for (int i = 0; i < 4; i++)
#pragma unroll
            for (int r = 0; r < 8; r++) {
                int bb = wm * 64 + i * 16 + hi * 8 + r;
                Zs[bb * 136 + wg * 32 + ul] = (__bf16)(acc[i][j][r] * sc + sh);
            }
    }
    __syncthreads();

    // fused gate math + cell BN (row 8); 16 (b,u) pairs per thread
    const int ul = tid & 31, b0 = tid >> 5;
    const int u = u0 + ul;
    const float sc8 = bn_gamma[8 * Uc + u] * rsqrtf(bn_var[8 * Uc + u] + BN_EPS);
    const float sh8 = bn_beta[8 * Uc + u] - bn_mean[8 * Uc + u] * sc8;
    const float* g_t = gates + (size_t)t * Bc * G4;
#pragma unroll 4
    for (int it = 0; it < 16; it++) {
        int bb = b0 + it * 8;
        float z0 = (float)Zs[bb * 136 + 0 * 32 + ul];
        float z1 = (float)Zs[bb * 136 + 1 * 32 + ul];
        float z2 = (float)Zs[bb * 136 + 2 * 32 + ul];
        float z3 = (float)Zs[bb * 136 + 3 * 32 + ul];
        const float* gb = g_t + (size_t)bb * G4;
        float xi = gb[u], xf = gb[Uc + u], xc = gb[2 * Uc + u], xo = gb[3 * Uc + u];
        float c  = cbuf[bb * Uc + u];
        float ig = hsig(xi + z0);
        float fg = hsig(xf + z1);
        float cn = fg * c + ig * fast_tanh(xc + z2);
        cn = cn * sc8 + sh8;                          // cell BN (row 8)
        float og = hsig(xo + z3);
        float h  = og * fast_tanh(cn);
        cbuf[bb * Uc + u] = cn;
        Hout[bb * Uc + u] = (__bf16)h;
        out[((size_t)bb * Tc + t) * Uc + u] = h;
    }
}

// ---------------------------------------------------------------- launch
extern "C" void kernel_launch(void* const* d_in, const int* in_sizes, int n_in,
                              void* d_out, int out_size, void* d_ws, size_t ws_size,
                              hipStream_t stream) {
    const float* x        = (const float*)d_in[0];
    const float* kernel   = (const float*)d_in[1];
    const float* rk       = (const float*)d_in[2];
    const float* bias     = (const float*)d_in[3];
    const float* bn_beta  = (const float*)d_in[4];
    const float* bn_gamma = (const float*)d_in[5];
    const float* bn_mean  = (const float*)d_in[6];
    const float* bn_var   = (const float*)d_in[7];
    float* out = (float*)d_out;

    char* ws = (char*)d_ws;
    size_t off = 0;
    __bf16* xb  = (__bf16*)(ws + off); off += (size_t)Bc * Tc * Dc * 2;   // 32 MB
    __bf16* wb  = (__bf16*)(ws + off); off += (size_t)Dc * G4 * 2;        //  4 MB
    __bf16* rkb = (__bf16*)(ws + off); off += (size_t)Uc * G4 * 2;        //  8 MB
    float*  gates = (float*)(ws + off); off += (size_t)Tc * Bc * G4 * 4;  // 512 MB
    float*  cbuf  = (float*)(ws + off); off += (size_t)Bc * Uc * 4;
    __bf16* h0  = (__bf16*)(ws + off); off += (size_t)Bc * Uc * 2;
    __bf16* h1  = (__bf16*)(ws + off); off += (size_t)Bc * Uc * 2;

    f32_to_bf16_kernel<<<4096, 256, 0, stream>>>(x, xb, Bc * Tc * Dc);
    f32_to_bf16_kernel<<<1024, 256, 0, stream>>>(kernel, wb, Dc * G4);
    f32_to_bf16_kernel<<<2048, 256, 0, stream>>>(rk, rkb, Uc * G4);
    init_state_kernel<<<256, 256, 0, stream>>>(cbuf, h0, Bc * Uc);

    gemm_input_kernel<<<dim3(Bc * Tc / 128, G4 / 128), 256, 0, stream>>>(
        xb, wb, bn_beta, bn_gamma, bn_mean, bn_var, bias, gates);

    for (int t = 0; t < Tc; t++) {
        const __bf16* hin = (t & 1) ? h1 : h0;
        __bf16* hout      = (t & 1) ? h0 : h1;
        lstm_step_kernel<<<Uc / 32, 256, 0, stream>>>(
            t, hin, rkb, bn_beta, bn_gamma, bn_mean, bn_var,
            gates, cbuf, hout, out);
    }
}
